// GNNModel_58385785422521
// MI455X (gfx1250) — compile-verified, matrix operand order checked by reading, not run on hardware
//
#include <hip/hip_runtime.h>
#include <hip/hip_bf16.h>

// ---------------------------------------------------------------------------
// GAT (3x GATConv + linear out) for gfx1250.
// Dense GEMMs -> v_wmma_f32_16x16x32_bf16 (f32 accumulate).
// Edge softmax/aggregation -> wave-per-edge gathers + global f32 atomics.
// ---------------------------------------------------------------------------

typedef __attribute__((ext_vector_type(16))) __bf16 v16bf;
typedef __attribute__((ext_vector_type(8)))  float  v8f;

union BfFrag {
    unsigned int u[8];
    v16bf bf;
};

#define HIDF 128

static __device__ __forceinline__ unsigned short f32_to_bf16(float f) {
    unsigned int u = __float_as_uint(f);
    // round-to-nearest-even truncation
    unsigned int r = u + 0x7FFFu + ((u >> 16) & 1u);
    return (unsigned short)(r >> 16);
}

static __device__ __forceinline__ void atomicMaxF(float* addr, float val) {
    // Monotonic-int trick: valid for all finite/inf floats.
    if (val >= 0.0f) atomicMax((int*)addr, __float_as_int(val));
    else             atomicMin((unsigned int*)addr, __float_as_uint(val));
}

// --------------------------- W -> Wt (bf16) --------------------------------
// Wt[n*K + k] = bf16(W[k*Nout + n]); makes the WMMA B fragment load pattern
// identical to the A fragment pattern (contiguous 16B runs per lane).
__global__ void wt_bf16_kernel(const float* __restrict__ W,
                               unsigned short* __restrict__ Wt,
                               int K, int Nout) {
    int i = blockIdx.x * blockDim.x + threadIdx.x;
    if (i < K * Nout) {
        int k = i / Nout, n = i - k * Nout;
        Wt[n * K + k] = f32_to_bf16(W[i]);
    }
}

// ------------------------------ WMMA GEMM ----------------------------------
// C[M,Nout] = A[M,K] @ W[K,Nout]  (+ bias), A f32, W given as bf16 Wt[Nout,K].
// blockDim.x = (Nout/16)*32. Each block: one 16-row M tile; wave w owns
// N-tile w. A tile staged f32->bf16 in LDS (shared by all waves).
__global__ void gemm_wmma_bf16(const float* __restrict__ A,
                               const unsigned short* __restrict__ Wt,
                               float* __restrict__ C,
                               const float* __restrict__ bias,
                               int M, int K, int Nout) {
    __shared__ unsigned short As[16 * (HIDF + 8)];
    const int ldA = K + 8;

    const int tid = threadIdx.x;
    const int m0  = blockIdx.x * 16;

    // Stage A tile (16 x K) into LDS as bf16, coalesced.
    const int total = 16 * K;
    for (int i = tid; i < total; i += blockDim.x) {
        int r = i / K, c = i - r * K;
        int gr = m0 + r;
        float v = (gr < M) ? A[(size_t)gr * K + c] : 0.0f;
        As[r * ldA + c] = f32_to_bf16(v);
    }
    __syncthreads();

    const int wave  = tid >> 5;
    const int lane  = tid & 31;
    const int row   = lane & 15;   // M (A) / N (B,C) within tile
    const int khalf = lane >> 4;   // lane-group K offset selector
    const int ncol  = wave * 16 + row;   // row of Wt == output column

    v8f acc = {};

    for (int k0 = 0; k0 < K; k0 += 32) {
        const int cb = k0 + khalf * 8;

        // A fragment: lane holds row, K = cb..cb+7 and cb+16..cb+23
        BfFrag a, b;
        {
            const uint4* p0 = (const uint4*)&As[row * ldA + cb];
            const uint4* p1 = (const uint4*)&As[row * ldA + cb + 16];
            uint4 x0 = *p0, x1 = *p1;
            a.u[0] = x0.x; a.u[1] = x0.y; a.u[2] = x0.z; a.u[3] = x0.w;
            a.u[4] = x1.x; a.u[5] = x1.y; a.u[6] = x1.z; a.u[7] = x1.w;
        }
        // B fragment from transposed weights (same pattern, row = out col)
        {
            const uint4* p0 = (const uint4*)&Wt[(size_t)ncol * K + cb];
            const uint4* p1 = (const uint4*)&Wt[(size_t)ncol * K + cb + 16];
            uint4 x0 = *p0, x1 = *p1;
            b.u[0] = x0.x; b.u[1] = x0.y; b.u[2] = x0.z; b.u[3] = x0.w;
            b.u[4] = x1.x; b.u[5] = x1.y; b.u[6] = x1.z; b.u[7] = x1.w;
        }

        acc = __builtin_amdgcn_wmma_f32_16x16x32_bf16(
            /*neg_a=*/false, a.bf, /*neg_b=*/false, b.bf,
            /*c_mod=*/(short)0, acc, /*reuse_a=*/false, /*reuse_b=*/false);
    }

    // C/D layout: lane holds column (row var), VGPR r -> M = r + khalf*8
    float badd = bias ? bias[ncol] : 0.0f;
#pragma unroll
    for (int r = 0; r < 8; ++r) {
        int gr = m0 + r + khalf * 8;
        if (gr < M) C[(size_t)gr * Nout + ncol] = acc[r] + badd;
    }
}

// -------------------------- per-node attention dots ------------------------
// s_src[n] = <h[n], a_src>, s_dst[n] = <h[n], a_dst>; one wave per node.
__global__ void node_scores_kernel(const float* __restrict__ h,
                                   const float* __restrict__ a_s,
                                   const float* __restrict__ a_d,
                                   float* __restrict__ s_src,
                                   float* __restrict__ s_dst, int N) {
    int gt   = blockIdx.x * blockDim.x + threadIdx.x;
    int node = gt >> 5;
    int lane = gt & 31;
    if (node >= N) return;
    const float* hp = h + (size_t)node * HIDF;
    float ss = 0.0f, sd = 0.0f;
#pragma unroll
    for (int f = lane; f < HIDF; f += 32) {
        float v = hp[f];
        ss += v * a_s[f];
        sd += v * a_d[f];
    }
#pragma unroll
    for (int off = 16; off; off >>= 1) {
        ss += __shfl_xor(ss, off, 32);
        sd += __shfl_xor(sd, off, 32);
    }
    if (lane == 0) { s_src[node] = ss; s_dst[node] = sd; }
}

// ------------------------------- init --------------------------------------
__global__ void init_kernel(float* __restrict__ m, float* __restrict__ denom,
                            float* __restrict__ hagg, int N) {
    int i = blockIdx.x * blockDim.x + threadIdx.x;
    if (i < N * HIDF) hagg[i] = 0.0f;
    if (i < N) { m[i] = __int_as_float(0xFF800000); denom[i] = 0.0f; }
}

static __device__ __forceinline__ void edge_ids(const int* __restrict__ ei,
                                                int i, int E, int& s, int& d) {
    if (i < E) { s = ei[i]; d = ei[E + i]; }
    else       { s = d = i - E; }   // appended self loops
}

// --------------------------- edge logits -> max ----------------------------
__global__ void edge_max_kernel(const int* __restrict__ ei,
                                const float* __restrict__ ssrc,
                                const float* __restrict__ sdst,
                                float* __restrict__ m, int E, int N) {
    int i = blockIdx.x * blockDim.x + threadIdx.x;
    if (i >= E + N) return;
    int s, d; edge_ids(ei, i, E, s, d);
    float e = ssrc[s] + sdst[d];
    e = e > 0.0f ? e : 0.2f * e;            // LeakyReLU(0.2)
    atomicMaxF(&m[d], e);
}

// ------------------------ edge exp + denom ---------------------------------
__global__ void edge_exp_kernel(const int* __restrict__ ei,
                                const float* __restrict__ ssrc,
                                const float* __restrict__ sdst,
                                const float* __restrict__ m,
                                float* __restrict__ ex,
                                float* __restrict__ denom, int E, int N) {
    int i = blockIdx.x * blockDim.x + threadIdx.x;
    if (i >= E + N) return;
    int s, d; edge_ids(ei, i, E, s, d);
    float e = ssrc[s] + sdst[d];
    e = e > 0.0f ? e : 0.2f * e;
    float v = __expf(e - m[d]);
    ex[i] = v;
    atomicAdd(&denom[d], v);
}

// ------------------- weighted scatter aggregation --------------------------
// One wave per edge; lane handles 4 consecutive features (128 = 32*4).
__global__ void aggregate_kernel(const int* __restrict__ ei,
                                 const float* __restrict__ ex,
                                 const float* __restrict__ denom,
                                 const float* __restrict__ h,
                                 float* __restrict__ hagg, int E, int N) {
    int gt   = blockIdx.x * blockDim.x + threadIdx.x;
    int edge = gt >> 5;
    int lane = gt & 31;
    if (edge >= E + N) return;
    int s, d; edge_ids(ei, edge, E, s, d);
    float alpha = ex[edge] / (denom[d] + 1e-16f);
    float4 v = ((const float4*)(h + (size_t)s * HIDF))[lane];
    float* out = hagg + (size_t)d * HIDF + lane * 4;
    atomicAdd(out + 0, alpha * v.x);
    atomicAdd(out + 1, alpha * v.y);
    atomicAdd(out + 2, alpha * v.z);
    atomicAdd(out + 3, alpha * v.w);
}

// ------------------------- bias + LeakyReLU(0.01) --------------------------
__global__ void bias_act_kernel(float* __restrict__ h,
                                const float* __restrict__ b, int N) {
    int i = blockIdx.x * blockDim.x + threadIdx.x;
    if (i < N * HIDF) {
        float v = h[i] + b[i & (HIDF - 1)];
        h[i] = v > 0.0f ? v : 0.01f * v;
    }
}

// ---------------------------------------------------------------------------
extern "C" void kernel_launch(void* const* d_in, const int* in_sizes, int n_in,
                              void* d_out, int out_size, void* d_ws, size_t ws_size,
                              hipStream_t stream) {
    (void)n_in; (void)out_size; (void)ws_size;

    const float* x    = (const float*)d_in[0];
    const int*   ei   = (const int*)d_in[1];
    const float* Wl[3] = {(const float*)d_in[2], (const float*)d_in[6],  (const float*)d_in[10]};
    const float* Asl[3]= {(const float*)d_in[3], (const float*)d_in[7],  (const float*)d_in[11]};
    const float* Adl[3]= {(const float*)d_in[4], (const float*)d_in[8],  (const float*)d_in[12]};
    const float* Bl[3] = {(const float*)d_in[5], (const float*)d_in[9],  (const float*)d_in[13]};
    const float* Wout = (const float*)d_in[14];
    const float* bout = (const float*)d_in[15];

    const int FIN  = 64;
    const int FOUT = 64;
    const int N  = in_sizes[0] / FIN;
    const int E  = in_sizes[1] / 2;
    const int Et = E + N;                      // with self loops

    // Workspace carve-up (floats)
    float* ws     = (float*)d_ws;
    float* h_pre  = ws;                                  // N*128
    float* h_agg  = h_pre + (size_t)N * HIDF;            // N*128
    float* s_src  = h_agg + (size_t)N * HIDF;            // N
    float* s_dst  = s_src + N;                           // N
    float* mbuf   = s_dst + N;                           // N
    float* dbuf   = mbuf  + N;                           // N
    float* exbuf  = dbuf  + N;                           // Et
    unsigned short* Wt = (unsigned short*)(exbuf + Et);  // <=128*128 bf16

    const int B256 = 256;
    const int gNH   = (N * HIDF + B256 - 1) / B256;
    const int gEt   = (Et + B256 - 1) / B256;
    const int gEtW  = (int)(((long long)Et * 32 + B256 - 1) / B256);
    const int gNW   = (int)(((long long)N * 32 + B256 - 1) / B256);
    const int gM    = (N + 15) / 16;

    const float* Ain = x;
    int Kin = FIN;
    for (int l = 0; l < 3; ++l) {
        wt_bf16_kernel<<<(Kin * HIDF + B256 - 1) / B256, B256, 0, stream>>>(
            Wl[l], Wt, Kin, HIDF);
        gemm_wmma_bf16<<<gM, (HIDF / 16) * 32, 0, stream>>>(
            Ain, Wt, h_pre, nullptr, N, Kin, HIDF);
        node_scores_kernel<<<gNW, B256, 0, stream>>>(
            h_pre, Asl[l], Adl[l], s_src, s_dst, N);
        init_kernel<<<gNH, B256, 0, stream>>>(mbuf, dbuf, h_agg, N);
        edge_max_kernel<<<gEt, B256, 0, stream>>>(ei, s_src, s_dst, mbuf, E, N);
        edge_exp_kernel<<<gEt, B256, 0, stream>>>(ei, s_src, s_dst, mbuf,
                                                  exbuf, dbuf, E, N);
        aggregate_kernel<<<gEtW, B256, 0, stream>>>(ei, exbuf, dbuf, h_pre,
                                                    h_agg, E, N);
        bias_act_kernel<<<gNH, B256, 0, stream>>>(h_agg, Bl[l], N);
        Ain = h_agg;
        Kin = HIDF;
    }

    // Output projection: out = h @ Wout + bout
    wt_bf16_kernel<<<(HIDF * FOUT + B256 - 1) / B256, B256, 0, stream>>>(
        Wout, Wt, HIDF, FOUT);
    gemm_wmma_bf16<<<gM, (FOUT / 16) * 32, 0, stream>>>(
        h_agg, Wt, (float*)d_out, bout, N, HIDF, FOUT);
}
